// decoder_80487687127254
// MI455X (gfx1250) — matrix operand order repaired and checked by reading, not compiled, not run on hardware
//
#include <hip/hip_runtime.h>
#include <hip/hip_fp16.h>

typedef __attribute__((ext_vector_type(16))) _Float16 v16h;
typedef __attribute__((ext_vector_type(8)))  float    v8f;

#define BM 128
#define BN 128
#define BK 32
#define APITCH 40   // 32 halfs + 8 pad (80B rows, 16B-aligned chunks)

// ---------------------------------------------------------------------------
// Async global->LDS helpers (CDNA5 GLOBAL_LOAD_ASYNC_TO_LDS_*, ASYNCcnt)
// ---------------------------------------------------------------------------
__device__ __forceinline__ void async_ld16(const void* gptr, void* lptr)
{
    uint32_t l = (uint32_t)(uintptr_t)lptr;           // LDS addr = low 32 bits
    uint64_t g = (uint64_t)(uintptr_t)gptr;
    asm volatile("global_load_async_to_lds_b128 %0, %1, off"
                 :: "v"(l), "v"(g) : "memory");
}
__device__ __forceinline__ void wait_async0()
{
    asm volatile("s_wait_asynccnt 0x0" ::: "memory");
}
__device__ __forceinline__ void wait_async4()
{
    asm volatile("s_wait_asynccnt 0x4" ::: "memory");
}

// ---------------------------------------------------------------------------
// f32 -> f16 conversion / transpose pre-passes (run once per launch)
// ---------------------------------------------------------------------------
__global__ void cvt_f16_kernel(const float* __restrict__ src,
                               _Float16* __restrict__ dst, int n)
{
    int i = blockIdx.x * blockDim.x + threadIdx.x;
    if (i < n) dst[i] = (_Float16)src[i];
}

// w (K,N) row-major f32 -> wt (N,K) row-major f16
__global__ void transpose_cvt_kernel(const float* __restrict__ w,
                                     _Float16* __restrict__ wt, int K, int N)
{
    int idx = blockIdx.x * blockDim.x + threadIdx.x;
    if (idx >= K * N) return;
    int k = idx / N, n = idx % N;
    wt[(size_t)n * K + k] = (_Float16)w[idx];
}

__global__ void zero_f32(float* __restrict__ p, int n)
{
    int i = blockIdx.x * blockDim.x + threadIdx.x;
    if (i < n) p[i] = 0.0f;
}

// ---------------------------------------------------------------------------
// Generic async-pipelined WMMA GEMM:
//   C = softsign(A[MxK] @ Bt[NxK]^T + bias)
// A,Bt f16 in global; double-buffered LDS tiles filled by async copies.
// M,N multiples of 128; K multiple of 32. F16OUT selects C dtype.
// ---------------------------------------------------------------------------
template <bool F16OUT>
__global__ __launch_bounds__(256) void gemm_async_wmma(
    const _Float16* __restrict__ A,   // (M,K)
    const _Float16* __restrict__ Bt,  // (N,K)  (= B transposed)
    const float* __restrict__ bias,   // (N)
    float* __restrict__ C32, _Float16* __restrict__ C16,
    int M, int N, int K)
{
    __shared__ _Float16 As[2][BM][APITCH];
    __shared__ _Float16 Bs[2][BN][APITCH];   // column-major tile: [n][k]

    const int tid   = threadIdx.x;
    const int lane  = tid & 31;
    const int wave  = tid >> 5;
    const int waveM = wave >> 2;   // 0..1
    const int waveN = wave & 3;    // 0..3
    const int m0 = blockIdx.y * BM;
    const int n0 = blockIdx.x * BN;
    const int row16 = lane & 15;
    const int half  = lane >> 4;

    v8f acc[4][2];
    #pragma unroll
    for (int mi = 0; mi < 4; ++mi)
        #pragma unroll
        for (int ni = 0; ni < 2; ++ni)
            #pragma unroll
            for (int e = 0; e < 8; ++e) acc[mi][ni][e] = 0.0f;

    // Issue one (A,B) tile pair: 4 async b128 copies per thread
    auto issueTile = [&](int k0, int buf) {
        #pragma unroll
        for (int c = 0; c < 2; ++c) {
            int chunk = tid + c * 256;       // 0..511
            int r  = chunk >> 2;             // 0..127
            int co = (chunk & 3) * 8;        // half offset 0,8,16,24
            async_ld16(A  + (size_t)(m0 + r) * K + k0 + co, &As[buf][r][co]);
            async_ld16(Bt + (size_t)(n0 + r) * K + k0 + co, &Bs[buf][r][co]);
        }
    };

    const int nk = K / BK;
    issueTile(0, 0);
    for (int kk = 0; kk < nk; ++kk) {
        const int buf = kk & 1;
        if (kk + 1 < nk) { issueTile((kk + 1) * BK, buf ^ 1); wait_async4(); }
        else             { wait_async0(); }
        __syncthreads();

        v16h bfr[2];
        #pragma unroll
        for (int ni = 0; ni < 2; ++ni) {
            const _Float16* bp = &Bs[buf][waveN * 32 + ni * 16 + row16][16 * half];
            #pragma unroll
            for (int e = 0; e < 16; ++e) bfr[ni][e] = bp[e];
        }
        v16h afr[4];
        #pragma unroll
        for (int mi = 0; mi < 4; ++mi) {
            const _Float16* ap = &As[buf][waveM * 64 + mi * 16 + row16][0];
            #pragma unroll
            for (int g = 0; g < 8; ++g) {
                int kk2 = 16 * half + 2 * g;
                afr[mi][2 * g]     = ap[kk2];
                afr[mi][2 * g + 1] = ap[kk2 + 1];
            }
        }
        #pragma unroll
        for (int mi = 0; mi < 4; ++mi)
            #pragma unroll
            for (int ni = 0; ni < 2; ++ni)
                acc[mi][ni] = __builtin_amdgcn_wmma_f32_16x16x32_f16(
                    false, afr[mi], false, bfr[ni],
                    (short)0, acc[mi][ni], false, false);
        __syncthreads();   // protect buf before it is refilled at kk+2
    }

    #pragma unroll
    for (int mi = 0; mi < 4; ++mi) {
        #pragma unroll
        for (int ni = 0; ni < 2; ++ni) {
            int col = n0 + waveN * 32 + ni * 16 + row16;
            float bv = bias[col];
            #pragma unroll
            for (int i = 0; i < 8; ++i) {
                int rowm = m0 + waveM * 64 + mi * 16 + half * 8 + i;
                float v = acc[mi][ni][i] + bv;
                v = v / (1.0f + fabsf(v));             // soft_sign
                if (F16OUT) C16[(size_t)rowm * N + col] = (_Float16)v;
                else        C32[(size_t)rowm * N + col] = v;
            }
        }
    }
}

// ---------------------------------------------------------------------------
// LSTM step GEMM (async pipeline):
//   gates[4096x2048] = [rep16[:,t,:], h16] @ [wihT; whhT]^T + bih + bhh
// Virtual K=768: k<256 from rep/wihT, else h/whhT (chunks never straddle).
// ---------------------------------------------------------------------------
__global__ __launch_bounds__(256) void lstm_gemm_wmma(
    const _Float16* __restrict__ rep16,  // (4096, 1792), slice at t*256
    const _Float16* __restrict__ h16,    // (4096, 512)
    const _Float16* __restrict__ wihT,   // (2048, 256)
    const _Float16* __restrict__ whhT,   // (2048, 512)
    const float* __restrict__ bih, const float* __restrict__ bhh,
    float* __restrict__ gates,           // (4096, 2048)
    int t)
{
    const int K = 768, N = 2048;
    __shared__ _Float16 As[2][BM][APITCH];
    __shared__ _Float16 Bs[2][BN][APITCH];

    const int tid   = threadIdx.x;
    const int lane  = tid & 31;
    const int wave  = tid >> 5;
    const int waveM = wave >> 2;
    const int waveN = wave & 3;
    const int m0 = blockIdx.y * BM;
    const int n0 = blockIdx.x * BN;
    const int row16 = lane & 15;
    const int half  = lane >> 4;

    v8f acc[4][2];
    #pragma unroll
    for (int mi = 0; mi < 4; ++mi)
        #pragma unroll
        for (int ni = 0; ni < 2; ++ni)
            #pragma unroll
            for (int e = 0; e < 8; ++e) acc[mi][ni][e] = 0.0f;

    auto issueTile = [&](int k0, int buf) {
        const bool fromX = (k0 < 256);
        #pragma unroll
        for (int c = 0; c < 2; ++c) {
            int chunk = tid + c * 256;
            int r  = chunk >> 2;
            int co = (chunk & 3) * 8;
            const _Float16* ga = fromX
                ? rep16 + (size_t)(m0 + r) * 1792 + t * 256 + k0 + co
                : h16   + (size_t)(m0 + r) * 512 + (k0 - 256) + co;
            async_ld16(ga, &As[buf][r][co]);
            const _Float16* gb = fromX
                ? wihT + (size_t)(n0 + r) * 256 + k0 + co
                : whhT + (size_t)(n0 + r) * 512 + (k0 - 256) + co;
            async_ld16(gb, &Bs[buf][r][co]);
        }
    };

    const int nk = K / BK;
    issueTile(0, 0);
    for (int kk = 0; kk < nk; ++kk) {
        const int buf = kk & 1;
        if (kk + 1 < nk) { issueTile((kk + 1) * BK, buf ^ 1); wait_async4(); }
        else             { wait_async0(); }
        __syncthreads();

        v16h bfr[2];
        #pragma unroll
        for (int ni = 0; ni < 2; ++ni) {
            const _Float16* bp = &Bs[buf][waveN * 32 + ni * 16 + row16][16 * half];
            #pragma unroll
            for (int e = 0; e < 16; ++e) bfr[ni][e] = bp[e];
        }
        v16h afr[4];
        #pragma unroll
        for (int mi = 0; mi < 4; ++mi) {
            const _Float16* ap = &As[buf][waveM * 64 + mi * 16 + row16][0];
            #pragma unroll
            for (int g = 0; g < 8; ++g) {
                int kk2 = 16 * half + 2 * g;
                afr[mi][2 * g]     = ap[kk2];
                afr[mi][2 * g + 1] = ap[kk2 + 1];
            }
        }
        #pragma unroll
        for (int mi = 0; mi < 4; ++mi)
            #pragma unroll
            for (int ni = 0; ni < 2; ++ni)
                acc[mi][ni] = __builtin_amdgcn_wmma_f32_16x16x32_f16(
                    false, afr[mi], false, bfr[ni],
                    (short)0, acc[mi][ni], false, false);
        __syncthreads();
    }

    #pragma unroll
    for (int mi = 0; mi < 4; ++mi) {
        #pragma unroll
        for (int ni = 0; ni < 2; ++ni) {
            int col = n0 + waveN * 32 + ni * 16 + row16;
            float bv = bih[col] + bhh[col];
            #pragma unroll
            for (int i = 0; i < 8; ++i) {
                int rowm = m0 + waveM * 64 + mi * 16 + half * 8 + i;
                gates[(size_t)rowm * N + col] = acc[mi][ni][i] + bv;
            }
        }
    }
}

// ---------------------------------------------------------------------------
// LSTM gate nonlinearity: updates c (f32, in-place), writes h as f16 for the
// next recurrence GEMM and as f32 into lstm1[:, t, dirOff:].
// ---------------------------------------------------------------------------
__global__ void lstm_pointwise(const float* __restrict__ gates,
                               float* __restrict__ c,
                               _Float16* __restrict__ hnext16,
                               float* __restrict__ lstm1,
                               int t, int dirOff)
{
    int idx = blockIdx.x * blockDim.x + threadIdx.x;   // 4096*512
    if (idx >= 4096 * 512) return;
    int b = idx >> 9;
    int j = idx & 511;
    const float* g = gates + (size_t)b * 2048;
    float gi = g[j], gf = g[512 + j], gg = g[1024 + j], go = g[1536 + j];
    float si = 1.0f / (1.0f + __expf(-gi));
    float sf = 1.0f / (1.0f + __expf(-gf));
    float so = 1.0f / (1.0f + __expf(-go));
    float cn = sf * c[idx] + si * tanhf(gg);
    float h  = so * tanhf(cn);
    c[idx] = cn;
    hnext16[idx] = (_Float16)h;
    lstm1[(size_t)b * 7168 + (size_t)t * 1024 + dirOff + j] = h;
}

__global__ void zero_f16(_Float16* __restrict__ p, int n)
{
    int i = blockIdx.x * blockDim.x + threadIdx.x;
    if (i < n) p[i] = (_Float16)0.0f;
}

// ---------------------------------------------------------------------------
// Attention: one workgroup per batch row; growing key set lives in LDS.
// Deterministic wave-shuffle reductions. Output stored f16 (feeds emb GEMM).
// ---------------------------------------------------------------------------
__global__ __launch_bounds__(256) void attention_kernel(
    const float* __restrict__ enc,    // (4096, 7, 1024)
    const float* __restrict__ dec,    // (4096, 7, 1024)
    _Float16* __restrict__ att16)     // (4096, 7, 1024)
{
    __shared__ float keys[13 * 1024];
    __shared__ float qs[1024];
    __shared__ float scores[16];

    const int b   = blockIdx.x;
    const int tid = threadIdx.x;
    const int wv  = tid >> 5;
    const int ln  = tid & 31;

    for (int i = tid; i < 7 * 1024; i += 256)
        keys[i] = enc[(size_t)b * 7168 + i];
    __syncthreads();

    for (int i = 0; i < 7; ++i) {
        const int klen = 7 + i;
        #pragma unroll
        for (int r = 0; r < 4; ++r) {
            int d = tid + 256 * r;
            qs[d] = dec[(size_t)b * 7168 + (size_t)i * 1024 + d];
        }
        __syncthreads();

        for (int tt = wv; tt < klen; tt += 8) {
            float p = 0.0f;
            for (int d = ln; d < 1024; d += 32)
                p += qs[d] * keys[tt * 1024 + d];
            #pragma unroll
            for (int off = 16; off > 0; off >>= 1)
                p += __shfl_down(p, off, 32);
            if (ln == 0) scores[tt] = p;
        }
        __syncthreads();

        if (tid == 0) {   // klen <= 13: serial, deterministic softmax
            float mx = scores[0];
            for (int tt = 1; tt < klen; ++tt) mx = fmaxf(mx, scores[tt]);
            float s = 0.0f;
            for (int tt = 0; tt < klen; ++tt) {
                float e = __expf(scores[tt] - mx);
                scores[tt] = e; s += e;
            }
            float inv = 1.0f / s;
            for (int tt = 0; tt < klen; ++tt) scores[tt] *= inv;
        }
        __syncthreads();

        #pragma unroll
        for (int r = 0; r < 4; ++r) {
            int d = tid + 256 * r;
            float ctx = 0.0f;
            for (int tt = 0; tt < klen; ++tt)
                ctx += scores[tt] * keys[tt * 1024 + d];
            float o = ctx + qs[d];
            att16[(size_t)b * 7168 + (size_t)i * 1024 + d] = (_Float16)o;
            keys[klen * 1024 + d] = o;
        }
        __syncthreads();
    }
}

// ---------------------------------------------------------------------------
// Fused transposed-conv chain + leaky_relu + softmax(axis=2) -> output.
// ---------------------------------------------------------------------------
__global__ __launch_bounds__(64) void deconv_softmax_kernel(
    const float* __restrict__ up,     // (28672, 1024) = softsign(emb)
    const float* __restrict__ dc3w, const float* __restrict__ dc3b,
    const float* __restrict__ dc1w, const float* __restrict__ dc1b,
    const float* __restrict__ dc2w, const float* __restrict__ dc2b,
    float* __restrict__ out)          // (4096, 7, 3, 24)
{
    __shared__ float x0[64 * 16];
    __shared__ float y3[32 * 18];
    __shared__ float y1[16 * 22];
    __shared__ float y2[3 * 24];

    const int bt  = blockIdx.x;
    const int tid = threadIdx.x;

    for (int i = tid; i < 1024; i += 64)
        x0[i] = up[(size_t)bt * 1024 + i];
    __syncthreads();

    for (int o = tid; o < 576; o += 64) {          // dc3: (64,16)->(32,18) k=1x3
        int co = o / 18, ow = o % 18;
        float acc = dc3b[co];
        for (int ci = 0; ci < 64; ++ci) {
            const float* w = &dc3w[ci * 96 + co * 3];
            #pragma unroll
            for (int k = 0; k < 3; ++k) {
                int iw = ow - k;
                if (iw >= 0 && iw < 16) acc += x0[ci * 16 + iw] * w[k];
            }
        }
        y3[o] = acc > 0.0f ? acc : 0.01f * acc;
    }
    __syncthreads();

    for (int o = tid; o < 352; o += 64) {          // dc1: (32,18)->(16,22) k=1x5
        int co = o / 22, ow = o % 22;
        float acc = dc1b[co];
        for (int ci = 0; ci < 32; ++ci) {
            const float* w = &dc1w[ci * 80 + co * 5];
            #pragma unroll
            for (int k = 0; k < 5; ++k) {
                int iw = ow - k;
                if (iw >= 0 && iw < 18) acc += y3[ci * 18 + iw] * w[k];
            }
        }
        y1[o] = acc > 0.0f ? acc : 0.01f * acc;
    }
    __syncthreads();

    for (int o = tid; o < 72; o += 64) {           // dc2: (16,1,22)->(1,3,24) k=3x3
        int oh = o / 24, ow = o % 24;
        float acc = dc2b[0];
        for (int ci = 0; ci < 16; ++ci) {
            const float* w = &dc2w[ci * 9 + oh * 3];
            #pragma unroll
            for (int k = 0; k < 3; ++k) {
                int iw = ow - k;
                if (iw >= 0 && iw < 22) acc += y1[ci * 22 + iw] * w[k];
            }
        }
        y2[o] = acc > 0.0f ? acc : 0.01f * acc;
    }
    __syncthreads();

    if (tid < 24) {                                // softmax over oh (3 vals)
        int ow = tid;
        float a0 = y2[ow], a1 = y2[24 + ow], a2 = y2[48 + ow];
        float m = fmaxf(a0, fmaxf(a1, a2));
        float e0 = __expf(a0 - m), e1 = __expf(a1 - m), e2 = __expf(a2 - m);
        float inv = 1.0f / (e0 + e1 + e2);
        size_t base = (size_t)bt * 72;
        out[base + ow]      = e0 * inv;
        out[base + 24 + ow] = e1 * inv;
        out[base + 48 + ow] = e2 * inv;
    }
}

// ---------------------------------------------------------------------------
extern "C" void kernel_launch(void* const* d_in, const int* in_sizes, int n_in,
                              void* d_out, int out_size, void* d_ws, size_t ws_size,
                              hipStream_t stream)
{
    (void)in_sizes; (void)n_in; (void)out_size; (void)ws_size;
    const float* enc     = (const float*)d_in[0];   // (4096,7,1024)
    const float* noise   = (const float*)d_in[1];   // (4096,256)
    const float* erep_w  = (const float*)d_in[2];   // (256,1792)
    const float* erep_b  = (const float*)d_in[3];
    const float* wih_f   = (const float*)d_in[4];   // (256,2048)
    const float* whh_f   = (const float*)d_in[5];   // (512,2048)
    const float* bih_f   = (const float*)d_in[6];
    const float* bhh_f   = (const float*)d_in[7];
    const float* wih_b   = (const float*)d_in[8];
    const float* whh_b   = (const float*)d_in[9];
    const float* bih_bb  = (const float*)d_in[10];
    const float* bhh_bb  = (const float*)d_in[11];
    const float* emb_w   = (const float*)d_in[12];  // (1024,1024)
    const float* emb_b   = (const float*)d_in[13];
    const float* dc3w    = (const float*)d_in[14];
    const float* dc3b    = (const float*)d_in[15];
    const float* dc1w    = (const float*)d_in[16];
    const float* dc1b    = (const float*)d_in[17];
    const float* dc2w    = (const float*)d_in[18];
    const float* dc2b    = (const float*)d_in[19];
    float* out = (float*)d_out;

    // ---- workspace carve-up ----
    float* gates = (float*)d_ws;                        // 4096*2048 f32
    float* cbuf  = gates + (size_t)4096 * 2048;         // 4096*512  f32
    float* lstm1 = cbuf  + (size_t)4096 * 512;          // 4096*7168 f32
    float* emb   = lstm1;                               // reuse after attention
    _Float16* f16base = (_Float16*)(lstm1 + (size_t)4096 * 7168);
    _Float16* noise16 = f16base;                        // 4096*256
    _Float16* rep16   = noise16 + (size_t)4096 * 256;   // 4096*1792
    _Float16* h16A    = rep16   + (size_t)4096 * 1792;  // 4096*512
    _Float16* h16B    = h16A    + (size_t)4096 * 512;   // 4096*512
    _Float16* att16   = h16B    + (size_t)4096 * 512;   // 28672*1024
    _Float16* erepT   = att16   + (size_t)28672 * 1024; // 1792*256
    _Float16* wihT_f  = erepT   + (size_t)1792 * 256;   // 2048*256
    _Float16* whhT_f  = wihT_f  + (size_t)2048 * 256;   // 2048*512
    _Float16* wihT_b  = whhT_f  + (size_t)2048 * 512;   // 2048*256
    _Float16* whhT_b  = wihT_b  + (size_t)2048 * 256;   // 2048*512
    _Float16* embT    = whhT_b  + (size_t)2048 * 512;   // 1024*1024

    dim3 blk(256);

    // 0) one-time f16 conversions / weight transposes
    cvt_f16_kernel<<<(4096 * 256) / 256, blk, 0, stream>>>(noise, noise16, 4096 * 256);
    transpose_cvt_kernel<<<(256 * 1792) / 256, blk, 0, stream>>>(erep_w, erepT, 256, 1792);
    transpose_cvt_kernel<<<(256 * 2048) / 256, blk, 0, stream>>>(wih_f, wihT_f, 256, 2048);
    transpose_cvt_kernel<<<(512 * 2048) / 256, blk, 0, stream>>>(whh_f, whhT_f, 512, 2048);
    transpose_cvt_kernel<<<(256 * 2048) / 256, blk, 0, stream>>>(wih_b, wihT_b, 256, 2048);
    transpose_cvt_kernel<<<(512 * 2048) / 256, blk, 0, stream>>>(whh_b, whhT_b, 512, 2048);
    transpose_cvt_kernel<<<(1024 * 1024) / 256, blk, 0, stream>>>(emb_w, embT, 1024, 1024);

    // 1) rep16 = softsign(noise @ embedrep_w + b)    M=4096 N=1792 K=256
    gemm_async_wmma<true><<<dim3(1792 / BN, 4096 / BM), blk, 0, stream>>>(
        noise16, erepT, erep_b, nullptr, rep16, 4096, 1792, 256);

    // 2) bidirectional LSTM (7 sequential steps per direction)
    const int nh = 4096 * 512;
    for (int dir = 0; dir < 2; ++dir) {
        const _Float16* wihT = dir ? wihT_b : wihT_f;
        const _Float16* whhT = dir ? whhT_b : whhT_f;
        const float* bih = dir ? bih_bb : bih_f;
        const float* bhh = dir ? bhh_bb : bhh_f;
        zero_f16<<<(nh + 255) / 256, blk, 0, stream>>>(h16A, nh);
        zero_f32<<<(nh + 255) / 256, blk, 0, stream>>>(cbuf, nh);
        _Float16* hp = h16A; _Float16* hn = h16B;
        for (int s = 0; s < 7; ++s) {
            int t = dir ? (6 - s) : s;
            lstm_gemm_wmma<<<dim3(2048 / BN, 4096 / BM), blk, 0, stream>>>(
                rep16, hp, wihT, whhT, bih, bhh, gates, t);
            lstm_pointwise<<<nh / 256, blk, 0, stream>>>(
                gates, cbuf, hn, lstm1, t, dir * 512);
            _Float16* tmp = hp; hp = hn; hn = tmp;
        }
    }

    // 3) attention (LDS-resident growing key set), f16 output
    attention_kernel<<<4096, 256, 0, stream>>>(enc, lstm1, att16);

    // 4) emb = softsign(att @ emb_w + emb_b)   M=28672 N=1024 K=1024, f32 out
    gemm_async_wmma<false><<<dim3(1024 / BN, 28672 / BM), blk, 0, stream>>>(
        att16, embT, emb_b, emb, nullptr, 28672, 1024, 1024);

    // 5) fused deconv chain + softmax -> out (4096,7,3,24)
    deconv_softmax_kernel<<<28672, 64, 0, stream>>>(
        emb, dc3w, dc3b, dc1w, dc1b, dc2w, dc2b, out);
}
